// FeatureEmbedding_67688684584979
// MI455X (gfx1250) — matrix operand order repaired and checked by reading, not compiled, not run on hardware
//
#include <hip/hip_runtime.h>

// ---------------- problem dims ----------------
#define Bdim   64
#define Sdim   4096
#define V1     30001
#define TDIM   128
#define DMODEL 256
#define FFDIM  512
#define EPSLN  1e-3f
#define RATE   0.1f

typedef __attribute__((ext_vector_type(16))) __bf16 v16bf;
typedef __attribute__((ext_vector_type(8)))  __bf16 v8bf;
typedef __attribute__((ext_vector_type(8)))  float  v8f;

// ---------------- workspace layout (bytes) ----------------
#define WS_CNT    0
#define WS_W1F    256
#define W1F_BYTES (64 * 512 * 2)       // 16 N-tiles * 4 K-chunks
#define WS_WPF    (WS_W1F + W1F_BYTES)
#define WPF_BYTES (256 * 512 * 2)      // 32 N-tiles * 8 K-chunks
#define WS_WLF    (WS_WPF + WPF_BYTES)

// ---------------- per-row feature count ----------------
__global__ __launch_bounds__(256) void fe_count(const int* __restrict__ ftok,
                                                int* __restrict__ cnt) {
  __shared__ float red[256];
  int b = blockIdx.x;
  float s = 0.f;
  for (int i = threadIdx.x; i < Sdim; i += 256)
    s += (ftok[b * Sdim + i] > 0) ? 1.f : 0.f;
  red[threadIdx.x] = s;
  __syncthreads();
  for (int off = 128; off; off >>= 1) {
    if (threadIdx.x < (unsigned)off) red[threadIdx.x] += red[threadIdx.x + off];
    __syncthreads();
  }
  if (threadIdx.x == 0) {
    int k = (int)(red[0] * RATE);            // tf cast truncates
    cnt[b] = k > Sdim ? Sdim : k;
  }
}

// ---------------- pack f32 weights into bf16 WMMA B-fragments ----------------
// B layout (16-bit, 32x16): N=lane&15; K = 16*(e>>3) + 8*half + (e&7).
// Fragment f = nt*kcs + kc; dst[f*512 + lane*16 + e]
__global__ __launch_bounds__(256) void fe_pack(const float* __restrict__ W,
                                               __bf16* __restrict__ dst,
                                               int K, int N) {
  int kcs   = K / 32;
  int frags = (N / 16) * kcs;
  int idx = blockIdx.x * 256 + threadIdx.x;
  if (idx >= frags * 32) return;
  int frag = idx >> 5, lane = idx & 31;
  int half = lane >> 4, ln = lane & 15;
  int n  = (frag / kcs) * 16 + ln;
  int k0 = (frag % kcs) * 32 + 8 * half;
  __bf16* d = dst + (size_t)idx * 16;
#pragma unroll
  for (int e = 0; e < 16; ++e)
    d[e] = (__bf16)W[(size_t)(k0 + e + (e & 8)) * N + n];
}

// A fragment (16-bit, 16x32) from a row-major LDS tile row:
// lane m = lane&15; elements e: K = 16*(e>>3) + 8*half + (e&7)
// -> two ds_load_b128 per fragment.
__device__ inline v16bf load_a_frag(const __bf16* row_base, int k0, int half) {
  const __bf16* p = row_base + k0 + 8 * half;
  v8bf lo = *(const v8bf*)p;
  v8bf hi = *(const v8bf*)(p + 16);
  return __builtin_shufflevector(lo, hi, 0, 1, 2, 3, 4, 5, 6, 7,
                                 8, 9, 10, 11, 12, 13, 14, 15);
}

// reduce across the 8 lanes sharing (ln & ~7); masks stay within a 16-lane half
__device__ inline float red8(float v) {
  v += __shfl_xor(v, 1, 32);
  v += __shfl_xor(v, 2, 32);
  v += __shfl_xor(v, 4, 32);
  return v;
}

// ---------------- fused main kernel: 64 tokens per workgroup ----------------
// 8 waves; weights read once per 64 tokens (4x less L2 traffic than 16-token
// tiles). LayerNorm stats computed from WMMA accumulators via shuffles + small
// LDS partial arrays; results normalized in registers and stored directly.
__global__ __launch_bounds__(256) void fe_main(
    const int* __restrict__ ftok, const float* __restrict__ rclr,
    const float* __restrict__ emb,
    const float* __restrict__ b1v, const float* __restrict__ g1,
    const float* __restrict__ be1,
    const float* __restrict__ bp, const float* __restrict__ gp,
    const float* __restrict__ bep,
    const float* __restrict__ bl, const float* __restrict__ gl,
    const float* __restrict__ bel,
    const int* __restrict__ cnt,
    const __bf16* __restrict__ w1f, const __bf16* __restrict__ wpf,
    const __bf16* __restrict__ wlf,
    float* __restrict__ o_mask, float* __restrict__ o_tok,
    float* __restrict__ o_emb, float* __restrict__ o_reg) {
  __shared__ __bf16 lxs[64][TDIM];     // 16 KB  activations bf16
  __shared__ __bf16 hb16[64][DMODEL];  // 32 KB  post-LN hidden bf16
  __shared__ float  redS[64][16];      // 4 KB   LN partial sums
  __shared__ float  redQ[64][16];      // 4 KB   LN partial sumsq
  __shared__ float  mstat[64], rstat[64];
  __shared__ int    stok[64];
  __shared__ float  srclr[64];

  const int tid  = threadIdx.x;
  const int b    = blockIdx.y;
  const int tile = blockIdx.x;                 // 64-token tile

  // ---- phase 0: tokens, mask, rclr ----
  if (tid < 64) {
    int r = tid;
    int s = tile * 64 + r;
    int ft = ftok[b * Sdim + s];
    int k  = cnt[b];
    unsigned h = (unsigned)(b * Sdim + s) * 2654435761u;
    h ^= h >> 16; h *= 2246822519u; h ^= h >> 13;
    int rnd  = (s >= Sdim - k) ? (int)(h % (unsigned)(V1 - 1)) + 1 : 0;
    int samp = ft + rnd;
    o_mask[(size_t)b * Sdim + s] = (ft != 0) ? 1.f : 0.f;
    o_tok [(size_t)b * Sdim + s] = (float)samp;
    int cidx = samp;
    if (cidx < 0) cidx = 0;
    if (cidx > V1 - 1) cidx = V1 - 1;          // jnp OOB indexing clips
    stok[r]  = cidx;
    srclr[r] = rclr[(size_t)b * Sdim + s] + (rnd > 0 ? 1.f : 0.f);
  }
  __syncthreads();

  // ---- gather embedding rows -> LDS bf16 (coalesced within rows) ----
#pragma unroll
  for (int j = 0; j < 32; ++j) {
    int e = tid + 256 * j;
    int row = e >> 7, col = e & 127;
    lxs[row][col] = (__bf16)emb[(size_t)stok[row] * TDIM + col];
  }
  __syncthreads();

  const int wave = tid >> 5;                   // wave32: 8 waves
  const int lane = tid & 31;
  const int half = lane >> 4;
  const int ln   = lane & 15;
  const int pidx = wave * 2 + (ln >> 3);       // partial slot 0..15
  const bool wr  = (ln & 7) == 0;

  // ---- phase 1: GEMM1 [64,128]x[128,256]; 4 M-tiles x 2 N-tiles per wave ----
  v8f a1[4][2];
#pragma unroll
  for (int mt = 0; mt < 4; ++mt) { a1[mt][0] = 0; a1[mt][1] = 0; }
#pragma unroll
  for (int kc = 0; kc < 4; ++kc) {
    int f0 = (wave * 2) * 4 + kc;
    int f1 = (wave * 2 + 1) * 4 + kc;
    v16bf bf0 = *(const v16bf*)(w1f + ((size_t)f0 * 32 + lane) * 16);
    v16bf bf1 = *(const v16bf*)(w1f + ((size_t)f1 * 32 + lane) * 16);
#pragma unroll
    for (int mt = 0; mt < 4; ++mt) {
      v16bf a = load_a_frag(&lxs[mt * 16 + ln][0], kc * 32, half);
      a1[mt][0] = __builtin_amdgcn_wmma_f32_16x16x32_bf16(
          false, a, false, bf0, (short)0, a1[mt][0], false, false);
      a1[mt][1] = __builtin_amdgcn_wmma_f32_16x16x32_bf16(
          false, a, false, bf1, (short)0, a1[mt][1], false, false);
    }
  }
  // bias + ReLU in registers, LN partials via shuffles
  {
    int n0 = wave * 32 + ln;
    float bb0 = b1v[n0], bb1 = b1v[n0 + 16];
#pragma unroll
    for (int mt = 0; mt < 4; ++mt) {
#pragma unroll
      for (int r = 0; r < 8; ++r) {
        float v0 = fmaxf(a1[mt][0][r] + bb0, 0.f);
        float v1 = fmaxf(a1[mt][1][r] + bb1, 0.f);
        a1[mt][0][r] = v0;
        a1[mt][1][r] = v1;
        float s  = red8(v0 + v1);
        float sq = red8(v0 * v0 + v1 * v1);
        int m = mt * 16 + r + 8 * half;
        if (wr) { redS[m][pidx] = s; redQ[m][pidx] = sq; }
      }
    }
  }
  __syncthreads();
  if (tid < 64) {
    float s = 0.f, ss = 0.f;
#pragma unroll
    for (int j = 0; j < 16; ++j) { s += redS[tid][j]; ss += redQ[tid][j]; }
    float m = s * (1.f / DMODEL);
    float var = ss * (1.f / DMODEL) - m * m;
    mstat[tid] = m;
    rstat[tid] = rsqrtf(var + EPSLN);
  }
  __syncthreads();
  // normalize + rclr scale -> bf16 LDS (A-operand for phase 2)
  {
    int n0 = wave * 32 + ln;
    float g0 = g1[n0], g16 = g1[n0 + 16];
    float e0 = be1[n0], e16 = be1[n0 + 16];
#pragma unroll
    for (int mt = 0; mt < 4; ++mt) {
#pragma unroll
      for (int r = 0; r < 8; ++r) {
        int m = mt * 16 + r + 8 * half;
        float mu = mstat[m], rs = rstat[m], sc = srclr[m];
        hb16[m][n0]      = (__bf16)(((a1[mt][0][r] - mu) * rs * g0  + e0)  * sc);
        hb16[m][n0 + 16] = (__bf16)(((a1[mt][1][r] - mu) * rs * g16 + e16) * sc);
      }
    }
  }
  __syncthreads();

  // ---- phase 2: two GEMMs [64,256]x[256,512] + ReLU + LN -> global ----
  size_t rowbase = (size_t)b * Sdim + (size_t)tile * 64;
  for (int mat = 0; mat < 2; ++mat) {
    const __bf16* Wfr  = (mat == 0) ? wlf : wpf;   // emb output (Wl) first
    const float*  bias = (mat == 0) ? bl  : bp;
    const float*  gam  = (mat == 0) ? gl  : gp;
    const float*  bet  = (mat == 0) ? bel : bep;
    float*        optr = (mat == 0) ? o_emb : o_reg;

    v8f acc[4][4];
#pragma unroll
    for (int mt = 0; mt < 4; ++mt)
#pragma unroll
      for (int t = 0; t < 4; ++t) acc[mt][t] = 0;

#pragma unroll
    for (int kc = 0; kc < 8; ++kc) {
      v16bf a[4];
#pragma unroll
      for (int mt = 0; mt < 4; ++mt)
        a[mt] = load_a_frag(&hb16[mt * 16 + ln][0], kc * 32, half);
#pragma unroll
      for (int t = 0; t < 4; ++t) {
        int frag = (wave * 4 + t) * 8 + kc;
        v16bf bf = *(const v16bf*)(Wfr + ((size_t)frag * 32 + lane) * 16);
#pragma unroll
        for (int mt = 0; mt < 4; ++mt)
          acc[mt][t] = __builtin_amdgcn_wmma_f32_16x16x32_bf16(
              false, a[mt], false, bf, (short)0, acc[mt][t], false, false);
      }
    }

    // bias + ReLU in registers, LN partials via shuffles
#pragma unroll
    for (int mt = 0; mt < 4; ++mt) {
#pragma unroll
      for (int r = 0; r < 8; ++r) {
        float s = 0.f, sq = 0.f;
#pragma unroll
        for (int t = 0; t < 4; ++t) {
          int n = (wave * 4 + t) * 16 + ln;
          float v = fmaxf(acc[mt][t][r] + bias[n], 0.f);
          acc[mt][t][r] = v;
          s += v; sq += v * v;
        }
        s  = red8(s);
        sq = red8(sq);
        int m = mt * 16 + r + 8 * half;
        if (wr) { redS[m][pidx] = s; redQ[m][pidx] = sq; }
      }
    }
    __syncthreads();
    if (tid < 64) {
      float s = 0.f, ss = 0.f;
#pragma unroll
      for (int j = 0; j < 16; ++j) { s += redS[tid][j]; ss += redQ[tid][j]; }
      float m = s * (1.f / FFDIM);
      float var = ss * (1.f / FFDIM) - m * m;
      mstat[tid] = m;
      rstat[tid] = rsqrtf(var + EPSLN);
    }
    __syncthreads();
    // normalize registers, store directly to HBM
#pragma unroll
    for (int mt = 0; mt < 4; ++mt) {
#pragma unroll
      for (int r = 0; r < 8; ++r) {
        int m = mt * 16 + r + 8 * half;
        float mu = mstat[m], rs = rstat[m];
        float* orow = optr + (rowbase + m) * FFDIM;
#pragma unroll
        for (int t = 0; t < 4; ++t) {
          int n = (wave * 4 + t) * 16 + ln;
          orow[n] = (acc[mt][t][r] - mu) * rs * gam[n] + bet[n];
        }
      }
    }
    __syncthreads();   // before redS/redQ/mstat reuse by next mat
  }
}

// ---------------- host-side launcher ----------------
extern "C" void kernel_launch(void* const* d_in, const int* in_sizes, int n_in,
                              void* d_out, int out_size, void* d_ws, size_t ws_size,
                              hipStream_t stream) {
  const int*   ftok = (const int*)d_in[0];
  const float* rclr = (const float*)d_in[1];
  const float* emb  = (const float*)d_in[2];
  const float* W1   = (const float*)d_in[3];
  const float* b1   = (const float*)d_in[4];
  const float* g1   = (const float*)d_in[5];
  const float* be1  = (const float*)d_in[6];
  const float* Wp   = (const float*)d_in[7];
  const float* bp   = (const float*)d_in[8];
  const float* gp   = (const float*)d_in[9];
  const float* bep  = (const float*)d_in[10];
  const float* Wl   = (const float*)d_in[11];
  const float* bl   = (const float*)d_in[12];
  const float* gl   = (const float*)d_in[13];
  const float* bel  = (const float*)d_in[14];

  char*   ws  = (char*)d_ws;
  int*    cnt = (int*)(ws + WS_CNT);
  __bf16* w1f = (__bf16*)(ws + WS_W1F);
  __bf16* wpf = (__bf16*)(ws + WS_WPF);
  __bf16* wlf = (__bf16*)(ws + WS_WLF);

  float* out    = (float*)d_out;
  float* o_mask = out;
  float* o_tok  = out + (size_t)Bdim * Sdim;
  float* o_emb  = out + 2 * (size_t)Bdim * Sdim;
  float* o_reg  = o_emb + (size_t)Bdim * Sdim * FFDIM;

  fe_count<<<Bdim, 256, 0, stream>>>(ftok, cnt);
  fe_pack<<<(64 * 32 + 255) / 256, 256, 0, stream>>>(W1, w1f, TDIM, DMODEL);
  fe_pack<<<(256 * 32 + 255) / 256, 256, 0, stream>>>(Wp, wpf, DMODEL, FFDIM);
  fe_pack<<<(256 * 32 + 255) / 256, 256, 0, stream>>>(Wl, wlf, DMODEL, FFDIM);
  fe_main<<<dim3(Sdim / 64, Bdim), 256, 0, stream>>>(
      ftok, rclr, emb, b1, g1, be1, bp, gp, bep, bl, gl, bel,
      cnt, w1f, wpf, wlf, o_mask, o_tok, o_emb, o_reg);
}
